// StarAttention_8701603741980
// MI455X (gfx1250) — matrix operand order repaired and checked by reading, not compiled, not run on hardware
//
#include <hip/hip_runtime.h>
#include <hip/hip_bf16.h>

typedef __attribute__((ext_vector_type(16))) _Float16 v16h;
typedef __attribute__((ext_vector_type(8)))  float    v8f;

#define HH      16
#define DD      64
#define SEQ     32768
#define BS      2048
#define NB      16
#define CHUNK   32
#define EPSF    1e-6f
#define TILE_WS 1056        /* 1024 A + 16 W + 16 G floats per (tile,split) */

// Q is pre-scaled by (1/sqrt(64)) * log2(e): scores leave the WMMA already in
// log2 domain, so softmax uses bare v_exp_f32 (exp2) with no per-element mul.
#define QSCALE  0.18033688011112042f   /* 0.125 * 1.4426950408889634 */

__device__ __forceinline__ v8f wmma16x16x32(v16h a, v16h b, v8f c) {
  // D = A(16x32 f16) * B(32x16 f16) + C(16x16 f32)
  return __builtin_amdgcn_wmma_f32_16x16x32_f16(
      /*neg_a=*/false, a, /*neg_b=*/false, b,
      /*c_mod=*/(short)0, c, /*reuse_a=*/false, /*reuse_b=*/false);
}

// One wave per block: computes per-split partial attention for a (head, q-tile).
__global__ __launch_bounds__(32)
void star_attn_partial(const float* __restrict__ ctx,   // [S, H, D] f32 (B=1)
                       const float* __restrict__ qry,   // [Q, H, D] f32
                       float* __restrict__ ws,
                       int nsplits) {
  __shared__ _Float16 ldsP[16 * 32];    // P tile bounce buffer (C-layout -> A-layout)

  const int lane = threadIdx.x;         // 0..31 (wave32)
  const int sp   = blockIdx.x % nsplits;
  const int tile = blockIdx.x / nsplits;   // 0..255
  const int h    = tile >> 4;
  const int qt   = tile & 15;
  const int half = (lane >= 16) ? 1 : 0;
  const int l16  = lane & 15;

  // ---------------- Q A-fragments (loaded once, scaled, f32 -> f16) ----------------
  // A k order used here (contiguous): lanes 0-15 hold M=lane, K=0..15; lanes 16-31 K=16..31
  const int qrow = qt * 16 + l16;
  const float* qp = qry + ((size_t)qrow * HH + h) * DD;
  const int adsel = half ? 16 : 0;
  v16h aQ0, aQ1;
  #pragma unroll
  for (int j = 0; j < 8; ++j) {
    aQ0[j]     = (_Float16)(qp[adsel + j] * QSCALE);
    aQ0[8 + j] = (_Float16)(qp[8 + adsel + j] * QSCALE);
    aQ1[j]     = (_Float16)(qp[32 + adsel + j] * QSCALE);
    aQ1[8 + j] = (_Float16)(qp[40 + adsel + j] * QSCALE);
  }

  // Split-level accumulators (rows: lanes<16 hold q=g, lanes>=16 hold q=g+8)
  float Grow[8], Wrow[8];
  v8f Aacc[4];
  #pragma unroll
  for (int g = 0; g < 8; ++g) { Grow[g] = -1e30f; Wrow[g] = 0.0f; }
  #pragma unroll
  for (int t = 0; t < 4; ++t)
    #pragma unroll
    for (int g = 0; g < 8; ++g) Aacc[t][g] = 0.0f;

  const int blocksPerSplit = NB / nsplits;
  const int key0 = sp * blocksPerSplit * BS;
  const int bdsel = half ? 16 : 0;   // B k order used here (matches A): lanes<16 K=0..15, lanes>=16 K=16..31

  for (int blk = 0; blk < blocksPerSplit; ++blk) {
    const int kb = key0 + blk * BS;
    // Per-block online-softmax state (log2 domain for max; linear for sums)
    float mrun[8], srun[8];
    v8f acc[4];
    #pragma unroll
    for (int g = 0; g < 8; ++g) { mrun[g] = -1e30f; srun[g] = 0.0f; }
    #pragma unroll
    for (int t = 0; t < 4; ++t)
      #pragma unroll
      for (int g = 0; g < 8; ++g) acc[t][g] = 0.0f;

    for (int ch = 0; ch < BS / CHUNK; ++ch) {
      const int kc = kb + ch * CHUNK;
      if (ch + 1 < BS / CHUNK)
        __builtin_prefetch(ctx + ((size_t)(kc + CHUNK + l16) * HH + h) * DD, 0, 0);

      // -------- S = Q * K^T : two 16x16 score tiles (keys kc..+15, kc+16..+31) --------
      v8f sc0, sc1;
      {
        const float* kr = ctx + ((size_t)(kc + l16) * HH + h) * DD;
        v16h b0, b1;
        #pragma unroll
        for (int j = 0; j < 16; ++j) {
          b0[j] = (_Float16)kr[bdsel + j];        // d = 0..31  (K of WMMA)
          b1[j] = (_Float16)kr[32 + bdsel + j];   // d = 32..63
        }
        v8f c;
        #pragma unroll
        for (int g = 0; g < 8; ++g) c[g] = 0.0f;
        c = wmma16x16x32(aQ0, b0, c);
        c = wmma16x16x32(aQ1, b1, c);
        sc0 = c;
      }
      {
        const float* kr = ctx + ((size_t)(kc + 16 + l16) * HH + h) * DD;
        v16h b0, b1;
        #pragma unroll
        for (int j = 0; j < 16; ++j) {
          b0[j] = (_Float16)kr[bdsel + j];
          b1[j] = (_Float16)kr[32 + bdsel + j];
        }
        v8f c;
        #pragma unroll
        for (int g = 0; g < 8; ++g) c[g] = 0.0f;
        c = wmma16x16x32(aQ0, b0, c);
        c = wmma16x16x32(aQ1, b1, c);
        sc1 = c;
      }

      // -------- online softmax over the 32-key chunk (scores are log2-scaled) --------
      float corr[8];
      #pragma unroll
      for (int g = 0; g < 8; ++g) {
        float cm = fmaxf(sc0[g], sc1[g]);
        cm = fmaxf(cm, __shfl_xor(cm, 1, 32));
        cm = fmaxf(cm, __shfl_xor(cm, 2, 32));
        cm = fmaxf(cm, __shfl_xor(cm, 4, 32));
        cm = fmaxf(cm, __shfl_xor(cm, 8, 32));   // row-max within 16-lane half
        float nm = fmaxf(mrun[g], cm);
        float cg = exp2f(mrun[g] - nm);
        float p0 = exp2f(sc0[g] - nm);
        float p1 = exp2f(sc1[g] - nm);
        float ss = p0 + p1;
        ss += __shfl_xor(ss, 1, 32);
        ss += __shfl_xor(ss, 2, 32);
        ss += __shfl_xor(ss, 4, 32);
        ss += __shfl_xor(ss, 8, 32);
        srun[g] = srun[g] * cg + ss;
        mrun[g] = nm;
        corr[g] = cg;
        const int qi = g + (half << 3);
        ldsP[qi * 32 + l16]      = (_Float16)p0;   // P(q, k=0..15)
        ldsP[qi * 32 + 16 + l16] = (_Float16)p1;   // P(q, k=16..31)
      }
      #pragma unroll
      for (int t = 0; t < 4; ++t)
        #pragma unroll
        for (int g = 0; g < 8; ++g) acc[t][g] *= corr[g];

      // P (16x32) as A-fragment from LDS (single wave: DS ops in-order, no barrier)
      v16h aP;
      {
        const int ks = half ? 16 : 0;
        #pragma unroll
        for (int j = 0; j < 8; ++j) {
          aP[j]     = ldsP[l16 * 32 + ks + j];
          aP[8 + j] = ldsP[l16 * 32 + 8 + ks + j];
        }
      }

      // -------- O += P * V : V chunk 32x64 as four 32x16 B-fragments --------
      const float* vb = ctx + ((size_t)(kc + (half ? 16 : 0)) * HH + h) * DD + l16;
      #pragma unroll
      for (int t = 0; t < 4; ++t) {
        v16h bV;
        #pragma unroll
        for (int j = 0; j < 16; ++j)
          bV[j] = (_Float16)vb[(size_t)j * (HH * DD) + t * 16];
        acc[t] = wmma16x16x32(aP, bV, acc[t]);
      }
    }

    // -------- fold block into split accumulators (exact reference weighting) --------
    // All max values are in log2 domain; differences are invariant, sums linear.
    float so[8], sn[8];
    #pragma unroll
    for (int g = 0; g < 8; ++g) {
      const float mb = mrun[g], sb = srun[g];
      const float Gn = fmaxf(Grow[g], mb);
      const float eg = exp2f(Grow[g] - Gn);
      const float em = exp2f(mb - Gn);
      Wrow[g] = Wrow[g] * eg + sb * em;              // w_b = exp(m_b - G) * s_b
      so[g] = eg;
      sn[g] = em * (sb / (sb + EPSF));               // out_b * w_b = acc_b*em*s_b/(s_b+eps)
      Grow[g] = Gn;
    }
    #pragma unroll
    for (int t = 0; t < 4; ++t)
      #pragma unroll
      for (int g = 0; g < 8; ++g)
        Aacc[t][g] = Aacc[t][g] * so[g] + acc[t][g] * sn[g];
  }

  // -------- write per-split partials: A[16][64], W[16], G[16] --------
  float* wsb = ws + (size_t)blockIdx.x * TILE_WS;  // blockIdx = tile*nsplits + sp
  #pragma unroll
  for (int t = 0; t < 4; ++t)
    #pragma unroll
    for (int g = 0; g < 8; ++g) {
      const int qi = g + (half << 3);
      const int d  = t * 16 + l16;
      wsb[qi * 64 + d] = Aacc[t][g];
    }
  if (l16 == 0) {
    #pragma unroll
    for (int g = 0; g < 8; ++g) {
      const int qi = g + (half << 3);
      wsb[1024 + qi] = Wrow[g];
      wsb[1040 + qi] = Grow[g];   // log2-domain max
    }
  }
}

// Cross-split LSE combine -> out[Q, H, D]  (G values are log2-domain)
__global__ __launch_bounds__(256)
void star_attn_combine(const float* __restrict__ ws, float* __restrict__ out,
                       int nsplits) {
  const int tile = blockIdx.x;           // 0..255
  const int h  = tile >> 4;
  const int qt = tile & 15;
  const int tid = threadIdx.x;
  const int d = tid & 63;
  const float* base = ws + (size_t)tile * nsplits * TILE_WS;
  for (int i = 0; i < 4; ++i) {
    const int q = (tid >> 6) + i * 4;    // 0..15
    float Gm = -1e30f;
    for (int sp = 0; sp < nsplits; ++sp)
      Gm = fmaxf(Gm, base[sp * TILE_WS + 1040 + q]);
    float W = 0.0f, A = 0.0f;
    for (int sp = 0; sp < nsplits; ++sp) {
      const float e = exp2f(base[sp * TILE_WS + 1040 + q] - Gm);
      W += e * base[sp * TILE_WS + 1024 + q];
      A += e * base[sp * TILE_WS + q * 64 + d];
    }
    out[((size_t)(qt * 16 + q) * HH + h) * DD + d] = A / (W + EPSF);
  }
}

extern "C" void kernel_launch(void* const* d_in, const int* in_sizes, int n_in,
                              void* d_out, int out_size, void* d_ws, size_t ws_size,
                              hipStream_t stream) {
  const float* ctx = (const float*)d_in[0];   // context [1,32768,16,64] f32
  const float* qry = (const float*)d_in[1];   // query   [1,256,16,64]  f32
  float* out = (float*)d_out;                 // [1,256,16,64] f32
  float* ws  = (float*)d_ws;

  int nsplits = 4;                             // keys split across waves for occupancy
  while (nsplits > 1 &&
         ws_size < (size_t)256 * nsplits * TILE_WS * sizeof(float))
    nsplits >>= 1;

  star_attn_partial<<<dim3(256 * nsplits), dim3(32), 0, stream>>>(ctx, qry, ws, nsplits);
  star_attn_combine<<<dim3(256), dim3(256), 0, stream>>>(ws, out, nsplits);
}
